// FAGRAGNet_22960895164952
// MI455X (gfx1250) — matrix operand order repaired and checked by reading, not compiled, not run on hardware
//
#include <hip/hip_runtime.h>
#include <hip/hip_bf16.h>

// ---------------------------------------------------------------------------
// FAGRAGNet forward for gfx1250 (MI455X). Memory-bound (~1GB traffic @23.3TB/s
// => ~40us floor). All dense matmuls run through v_wmma_f32_16x16x32_f16.
// M/N edges handled by index CLAMPING only (dead rows/cols never stored);
// K edge handled in one masked step. Main K-loop is branch-free on A and
// uses walking pointers (no per-iteration 64-bit address math).
// ---------------------------------------------------------------------------

typedef __attribute__((ext_vector_type(16))) _Float16 v16h;
typedef __attribute__((ext_vector_type(8)))  _Float16 v8h;
typedef __attribute__((ext_vector_type(2)))  _Float16 v2h;
typedef __attribute__((ext_vector_type(8)))  float    v8f;

#define ACT_NONE 0
#define ACT_RELU 1
#define ACT_TANH 2

// ---------------------------------------------------------------------------
// WMMA GEMM:  C = act(A(MxK) * B(KxN) + bias + add) .* mul + res
// Row-major f32 operands; f16 conversion at LDS stage; f32 accumulate.
// TILE x TILE block tile, 4 waves in 2x2; each wave owns (TILE/2)^2 as
// (TILE/32)x(TILE/32) 16x16 WMMA sub-tiles. K-step 32.
// Batched via blockIdx.z with strides sA/sB/sC (add/mul/res follow sC).
// ---------------------------------------------------------------------------
template <int TILE>
__global__ __launch_bounds__(128) void gemm_wmma(
    const float* __restrict__ A, const float* __restrict__ B,
    float* __restrict__ C,
    const float* __restrict__ bias, const float* __restrict__ add,
    const float* __restrict__ mul, const float* __restrict__ res,
    int M, int N, int K,
    long long sA, long long sB, long long sC, int act)
{
  constexpr int S = TILE / 32;         // WMMA sub-tiles per wave per dim
  const int lda = K, ldb = N, ldc = N;
  const long long bz = blockIdx.z;
  A += bz * sA;  B += bz * sB;  C += bz * sC;
  if (add) add += bz * sC;
  if (mul) mul += bz * sC;
  if (res) res += bz * sC;

  __shared__ alignas(16) _Float16 As[TILE * 40];  // row-major, stride 40
  __shared__ alignas(16) _Float16 Bs[TILE * 40];  // col-major (Bs[col][k])

  const int tid  = threadIdx.x;
  const int bM   = blockIdx.y * TILE;
  const int bN   = blockIdx.x * TILE;
  const int w    = tid >> 5;
  const int lane = tid & 31;
  const int wm   = w >> 1, wn = w & 1;
  const int hl   = lane >> 4;          // lane half
  const int r    = lane & 15;

  v8f acc[S][S];
  const v8f vzero = {0.f, 0.f, 0.f, 0.f, 0.f, 0.f, 0.f, 0.f};
#pragma unroll
  for (int mt = 0; mt < S; ++mt)
#pragma unroll
    for (int nt = 0; nt < S; ++nt) acc[mt][nt] = vzero;

  // ---- loop-invariant staging geometry (row/col CLAMPED, never masked) ----
  // A: S chunks of (1 row x 8 k) per thread
  const float* aptr[S];
  int adst[S], aseg[S];
#pragma unroll
  for (int i = 0; i < S; ++i) {
    int c    = tid + i * 128;
    int row  = c >> 2, seg = (c & 3) * 8;
    int arow = bM + row;
    int rc   = arow < M ? arow : M - 1;            // clamp: dead rows harmless
    aptr[i]  = A + (long long)rc * lda + seg;
    adst[i]  = row * 40 + seg;
    aseg[i]  = seg;
  }
  // B: one chunk of (2 k x 8 cols) per thread (threads < TILE*2 active)
  const bool bact  = (tid < TILE * 2);
  const int  cs    = tid % (TILE / 8);
  const int  kp    = tid / (TILE / 8);
  const int  col0  = bN + cs * 8;
  const bool bfull = (col0 + 8 <= N);              // loop-invariant fast path
  int bcol[8];
#pragma unroll
  for (int e = 0; e < 8; ++e) {
    int gcol = col0 + e;
    bcol[e] = gcol < N ? gcol : N - 1;             // clamp: dead cols harmless
  }
  const int bdst = cs * 320 + 2 * kp;              // Bs index base (halfs)

  // ---- fragment load + 4(or 1) WMMAs ----
  auto compute = [&]() {
    v16h afr[S], bfr[S];
#pragma unroll
    for (int mt = 0; mt < S; ++mt) {
      int rb = (wm * (TILE / 2) + mt * 16 + r) * 40;
      v8h alo = *(const v8h*)(As + rb + hl * 8);
      v8h ahi = *(const v8h*)(As + rb + 16 + hl * 8);
      afr[mt] = __builtin_shufflevector(alo, ahi, 0,1,2,3,4,5,6,7,8,9,10,11,12,13,14,15);
    }
#pragma unroll
    for (int nt = 0; nt < S; ++nt) {
      int cb = (wn * (TILE / 2) + nt * 16 + r) * 40;
      v8h blo = *(const v8h*)(Bs + cb + hl * 8);
      v8h bhi = *(const v8h*)(Bs + cb + 16 + hl * 8);
      bfr[nt] = __builtin_shufflevector(blo, bhi, 0,1,2,3,4,5,6,7,8,9,10,11,12,13,14,15);
    }
#pragma unroll
    for (int mt = 0; mt < S; ++mt)
#pragma unroll
      for (int nt = 0; nt < S; ++nt)
        acc[mt][nt] = __builtin_amdgcn_wmma_f32_16x16x32_f16(
            false, afr[mt], false, bfr[nt], (short)0, acc[mt][nt], false, false);
  };

  // ---- main loop: full 32-wide K steps, walking pointers ----
  const int kmain = K & ~31;
  const float* apk[S];
#pragma unroll
  for (int i = 0; i < S; ++i) apk[i] = aptr[i];
  const float* brow = B + (long long)(2 * kp) * ldb;   // walking B row (k = 2kp)

  for (int kk = 0; kk < kmain; kk += 32) {
    __syncthreads();
#pragma unroll
    for (int i = 0; i < S; ++i) {
      const float* ap = apk[i];
      v8h h8;
#pragma unroll
      for (int e = 0; e < 8; ++e) h8[e] = (_Float16)ap[e];
      *(v8h*)(As + adst[i]) = h8;
      apk[i] += 32;
    }
    if (bact) {
      const float* b0 = brow;
      const float* b1 = brow + ldb;
      if (bfull) {                                 // contiguous: vectorizable
#pragma unroll
        for (int e = 0; e < 8; ++e) {
          v2h p; p[0] = (_Float16)b0[col0 + e]; p[1] = (_Float16)b1[col0 + e];
          *(v2h*)(Bs + bdst + e * 40) = p;
        }
      } else {                                     // N-edge: clamped indirect
#pragma unroll
        for (int e = 0; e < 8; ++e) {
          v2h p; p[0] = (_Float16)b0[bcol[e]]; p[1] = (_Float16)b1[bcol[e]];
          *(v2h*)(Bs + bdst + e * 40) = p;
        }
      }
      brow += (long long)32 * ldb;
    }
    __syncthreads();
    compute();
  }

  // ---- K remainder (only K=265, K=62 here): clamp + zero-mask on k ----
  if (kmain < K) {
    __syncthreads();
#pragma unroll
    for (int i = 0; i < S; ++i) {
      v8h h8;
#pragma unroll
      for (int e = 0; e < 8; ++e) {
        int kc  = kmain + aseg[i] + e;
        int kcl = kc < K ? kc : K - 1;
        float va = aptr[i][kcl - aseg[i]];         // stays inside the row
        h8[e] = (_Float16)(kc < K ? va : 0.f);
      }
      *(v8h*)(As + adst[i]) = h8;
    }
    if (bact) {
      int k0 = kmain + 2 * kp, k1 = k0 + 1;
      int k0c = k0 < K ? k0 : K - 1;
      int k1c = k1 < K ? k1 : K - 1;
      const float* b0 = B + (long long)k0c * ldb;
      const float* b1 = B + (long long)k1c * ldb;
#pragma unroll
      for (int e = 0; e < 8; ++e) {
        float f0 = b0[bcol[e]], f1 = b1[bcol[e]];
        v2h p;
        p[0] = (_Float16)(k0 < K ? f0 : 0.f);
        p[1] = (_Float16)(k1 < K ? f1 : 0.f);
        *(v2h*)(Bs + bdst + e * 40) = p;
      }
    }
    __syncthreads();
    compute();
  }

  // ---- epilogue: C VGPR rr -> (M = rr + hl*8, N = r) ----
#pragma unroll
  for (int mt = 0; mt < S; ++mt) {
#pragma unroll
    for (int nt = 0; nt < S; ++nt) {
      int gn = bN + wn * (TILE / 2) + nt * 16 + r;
#pragma unroll
      for (int rr = 0; rr < 8; ++rr) {
        int gm = bM + wm * (TILE / 2) + mt * 16 + hl * 8 + rr;
        if (gm < M && gn < N) {
          float val = acc[mt][nt][rr];
          if (bias) val += bias[gn];
          if (add)  val += add[(long long)gm * ldc + gn];
          if (act == ACT_RELU)      val = fmaxf(val, 0.f);
          else if (act == ACT_TANH) val = tanhf(val);
          if (mul) val *= mul[(long long)gm * ldc + gn];
          if (res) val += res[(long long)gm * ldc + gn];
          C[(long long)gm * ldc + gn] = val;
        }
      }
    }
  }
}

// ---------------------------------------------------------------------------
// MHSA pieces (8 heads, 5 tokens, d=64 — tiny, VALU)
// ---------------------------------------------------------------------------
__global__ void k_attn_scores(const float* __restrict__ q, const float* __restrict__ k,
                              float* __restrict__ dist, int total)
{
  int idx = blockIdx.x * blockDim.x + threadIdx.x;
  if (idx >= total) return;                       // idx = (n*8+h)*5+qi
  int qi = idx % 5, h = (idx / 5) % 8, n = idx / 40;
  const float* qp = q + ((long long)(n * 5 + qi)) * 512 + h * 64;
  float sc[5], mx = -3.4e38f;
  for (int kj = 0; kj < 5; ++kj) {
    const float* kp = k + ((long long)(n * 5 + kj)) * 512 + h * 64;
    float d = 0.f;
    for (int e = 0; e < 64; ++e) d += qp[e] * kp[e];
    d *= 0.125f;                                   // 1/sqrt(64)
    sc[kj] = d; mx = fmaxf(mx, d);
  }
  float s = 0.f;
  for (int kj = 0; kj < 5; ++kj) { sc[kj] = __expf(sc[kj] - mx); s += sc[kj]; }
  float inv = 1.f / s;
  float* dp = dist + (long long)idx * 5;
  for (int kj = 0; kj < 5; ++kj) dp[kj] = sc[kj] * inv;
}

__global__ void k_attn_apply(const float* __restrict__ dist, const float* __restrict__ v,
                             float* __restrict__ att, int total)
{
  int idx = blockIdx.x * blockDim.x + threadIdx.x;
  if (idx >= total) return;                       // (n,h,qi,d)
  int d  = idx & 63;
  int qi = (idx >> 6) % 5;
  int h  = (idx / 320) & 7;
  int n  = idx / 2560;
  const float* dp = dist + ((long long)((n * 8 + h) * 5 + qi)) * 5;
  float a = 0.f;
  for (int kj = 0; kj < 5; ++kj)
    a += dp[kj] * v[((long long)(n * 5 + kj)) * 512 + h * 64 + d];
  // transposed layout (N, qi, h, d) = (N, 2560) for the wo GEMM
  att[(long long)n * 2560 + qi * 512 + h * 64 + d] = a;
}

__global__ void k_band(const float* __restrict__ s, const float* __restrict__ cw,
                       const float* __restrict__ cb, float* __restrict__ wout, int N)
{
  int n = blockIdx.x * blockDim.x + threadIdx.x;
  if (n >= N) return;
  float t[5], mx = -3.4e38f;
  for (int j = 0; j < 5; ++j) {
    float a = cb[j];
    for (int i = 0; i < 5; ++i) a += s[n * 5 + i] * cw[i * 5 + j];
    t[j] = tanhf(a); mx = fmaxf(mx, t[j]);
  }
  float sum = 0.f;
  for (int j = 0; j < 5; ++j) { t[j] = __expf(t[j] - mx); sum += t[j]; }
  float inv = 1.f / sum;
  for (int j = 0; j < 5; ++j) wout[n * 5 + j] = t[j] * inv;
}

__global__ void k_scale(const float* __restrict__ x, const float* __restrict__ wb,
                        float* __restrict__ xs, int total)
{
  int idx = blockIdx.x * blockDim.x + threadIdx.x;
  if (idx >= total) return;
  int band = (idx / 265) % 5;
  int n    = idx / 1325;
  xs[idx] = x[idx] * wb[n * 5 + band];
}

// ---------------------------------------------------------------------------
// Fused conv2d + ReLU + maxpool (pool along width, -inf padding semantics)
// ---------------------------------------------------------------------------
__global__ void k_convpool(const float* __restrict__ in, const float* __restrict__ w,
                           const float* __restrict__ bi, float* __restrict__ out,
                           int Nn, int Cin, int Cout, int H, int Win,
                           int KH, int KW, int padH, int padW, int Wconv,
                           int pool, int ppad, int Wout)
{
  long long idx = (long long)blockIdx.x * blockDim.x + threadIdx.x;
  long long total = (long long)Nn * Cout * H * Wout;
  if (idx >= total) return;
  int ow = (int)(idx % Wout);
  int h  = (int)((idx / Wout) % H);
  int co = (int)((idx / ((long long)Wout * H)) % Cout);
  long long n = idx / ((long long)Wout * H * Cout);

  float m = -3.4e38f;
  for (int wi = 0; wi < pool; ++wi) {
    int cw = ow * pool - ppad + wi;                // conv-output column
    if (cw < 0 || cw >= Wconv) continue;
    float acc = bi[co];
    for (int ci = 0; ci < Cin; ++ci)
      for (int kh = 0; kh < KH; ++kh) {
        int ih = h - padH + kh;
        if (ih < 0 || ih >= H) continue;
        for (int kw = 0; kw < KW; ++kw) {
          int iw = cw - padW + kw;
          if (iw < 0 || iw >= Win) continue;
          acc += in[(((long long)n * Cin + ci) * H + ih) * Win + iw] *
                 w[(((long long)co * Cin + ci) * KH + kh) * KW + kw];
        }
      }
    acc = fmaxf(acc, 0.f);                         // ReLU then pool
    m = fmaxf(m, acc);
  }
  out[idx] = m;
}

// ---------------------------------------------------------------------------
// Graph adjacency: scores (xa xa^T), row softmax + top-10 sparsification
// ---------------------------------------------------------------------------
__global__ void k_adj_scores(const float* __restrict__ xa, float* __restrict__ adj, int total)
{
  int idx = blockIdx.x * blockDim.x + threadIdx.x;
  if (idx >= total) return;                       // (b,i,j), 32*62*62
  int j = idx % 62, i = (idx / 62) % 62, b = idx / 3844;
  const float* xi = xa + ((long long)(b * 62 + i)) * 64;
  const float* xj = xa + ((long long)(b * 62 + j)) * 64;
  float d = 0.f;
  for (int e = 0; e < 64; ++e) d += xi[e] * xj[e];
  adj[idx] = d;
}

__global__ void k_adj_sm_topk(float* __restrict__ adj, int rows)
{
  int idx = blockIdx.x * blockDim.x + threadIdx.x;
  if (idx >= rows) return;                        // one row (b,i)
  float* row = adj + (long long)idx * 62;
  float v[62];
  float mx = -3.4e38f;
  for (int j = 0; j < 62; ++j) { v[j] = row[j]; mx = fmaxf(mx, v[j]); }
  float s = 0.f;
  for (int j = 0; j < 62; ++j) { v[j] = __expf(v[j] - mx); s += v[j]; }
  float inv = 1.f / s;
  for (int j = 0; j < 62; ++j) v[j] *= inv;
  unsigned long long sel = 0ULL;                  // top-10, first-index tiebreak
  for (int t = 0; t < 10; ++t) {
    int best = 0; float bv = -3.4e38f;
    for (int j = 0; j < 62; ++j)
      if (!((sel >> j) & 1ULL) && v[j] > bv) { bv = v[j]; best = j; }
    sel |= 1ULL << best;
  }
  for (int j = 0; j < 62; ++j) row[j] = ((sel >> j) & 1ULL) ? v[j] : 0.f;
}

// concat [x1..x5] along node axis: cat(b, blk*62+i, c)
__global__ void k_concat(const float* __restrict__ xk, float* __restrict__ cat, int total)
{
  int idx = blockIdx.x * blockDim.x + threadIdx.x;
  if (idx >= total) return;                       // 32*5*62*32
  int c   = idx & 31;
  int i   = (idx >> 5) % 62;
  int blk = (idx / (32 * 62)) % 5;
  int b   = idx / (32 * 62 * 5);
  cat[(((long long)(b * 5 + blk) * 62) + i) * 32 + c] =
      xk[(long long)blk * 63488 + ((long long)(b * 62 + i)) * 32 + c];
}

// ---------------------------------------------------------------------------
static inline void launch_gemm(hipStream_t st,
    const float* A, const float* B, float* C,
    const float* bias, const float* add, const float* mul, const float* res,
    int M, int N, int K, int act,
    int batch = 1, long long sA = 0, long long sB = 0, long long sC = 0)
{
  if (M >= 48 && N >= 48) {
    dim3 g((N + 63) / 64, (M + 63) / 64, batch);
    gemm_wmma<64><<<g, 128, 0, st>>>(A, B, C, bias, add, mul, res,
                                     M, N, K, sA, sB, sC, act);
  } else {
    dim3 g((N + 31) / 32, (M + 31) / 32, batch);
    gemm_wmma<32><<<g, 128, 0, st>>>(A, B, C, bias, add, mul, res,
                                     M, N, K, sA, sB, sC, act);
  }
}

extern "C" void kernel_launch(void* const* d_in, const int* in_sizes, int n_in,
                              void* d_out, int out_size, void* d_ws, size_t ws_size,
                              hipStream_t stream)
{
  (void)in_sizes; (void)out_size; (void)ws_size;
  if (n_in < 149) return;   // expect 149 leaves (jax tree_flatten alphabetical)

  auto F = [&](int i) -> const float* { return (const float*)d_in[i]; };
  // params leaves, alphabetical dict-key order; 'x' is the last leaf.
  const float *c1_b = F(0),  *c1_w = F(1),  *c2_b = F(2),  *c2_w = F(3);
  const float *c3_b = F(4),  *c3_w = F(5),  *cwa_b = F(6), *cwa_w = F(7);
  const float *gate_b = F(8), *gate_w = F(9);
  const float *lin1_b = F(10), *lin1_w = F(11), *lin2_b = F(12), *lin2_w = F(13);
  const float *mc1_b = F(14), *mc1_w = F(15), *mc2_b = F(16), *mc2_w = F(17);
  const float *bo = F(18), *wk = F(19), *wo = F(20), *wq = F(21), *wv = F(22);
  const float *x = F(148);
  // sogc[g] leaves at base 23 + 25*g:
  //   0=bnlin_b 1=bnlin_w, convs[j]: 2+3j=rel_b 3+3j=rel_w 4+3j=root_w, 23=gate_b 24=gate_w
  auto SG = [&](int g, int leaf) -> const float* { return F(23 + 25 * g + leaf); };

  // -------- workspace layout (floats) --------
  float* W0 = (float*)d_ws;
  size_t off = 0;
  auto alloc = [&](size_t n) { float* p = W0 + off; off += (n + 63) & ~(size_t)63; return p; };
  float* xs    = alloc(2628800);                  // (1984,5,265) scaled input
  float* q     = alloc((size_t)9920 * 512);
  float* kb    = alloc((size_t)9920 * 512);
  float* vb    = alloc((size_t)9920 * 512);
  float* dist  = alloc(1984 * 8 * 25);
  float* att   = alloc((size_t)1984 * 2560);
  float* sb    = alloc(1984 * 5);
  float* wband = alloc(1984 * 5);
  float* p1    = alloc((size_t)1984 * 32 * 5 * 66);   // after mc1+pool4
  float* p2    = alloc((size_t)1984 * 32 * 5 * 17);   // after mc2+pool4(pad2)
  float* p3    = alloc((size_t)1984 * 32 * 5 * 7);    // after c1+pool2
  float* p4    = alloc((size_t)1984 * 64 * 5 * 2);    // after c2+pool2
  float* p5    = alloc((size_t)1984 * 128 * 5);       // after c3
  float* xa    = alloc(1984 * 64);
  float* adjm  = alloc(32 * 62 * 62);
  float* tbuf  = alloc(1984 * 64);
  float* hbuf  = alloc(1984 * 64);
  float* h2buf = alloc(1984 * 64);
  float* gmask = alloc(1984 * 32);
  float* xk    = alloc((size_t)5 * 1984 * 32);
  float* cat   = alloc((size_t)32 * 9920);
  float* gated = alloc((size_t)32 * 9920);
  float* l1    = alloc(32 * 1024);
  // adj@x scratch (max 1984*10560 = 20.95M floats) aliases the xs..wband
  // region (23.3M floats), which is dead once the conv frontend starts.
  float* adjx = xs;

  // -------- residual graph block --------
  auto graph = [&](const float* xin, int in_f, int g, float* out_slice) {
    launch_gemm(stream, xin, SG(g, 24), gmask, SG(g, 23), nullptr, nullptr, nullptr,
                1984, 32, in_f, ACT_TANH);                           // x_mask = tanh(x gate_w + b)
    launch_gemm(stream, xin, SG(g, 1), xa, SG(g, 0), nullptr, nullptr, nullptr,
                1984, 64, in_f, ACT_TANH);                           // xa = tanh(x bnlin + b)
    { int t = 32 * 62 * 62;
      k_adj_scores<<<(t + 127) / 128, 128, 0, stream>>>(xa, adjm, t); }
    k_adj_sm_topk<<<(1984 + 63) / 64, 64, 0, stream>>>(adjm, 1984);
    // adjx = adj @ x   (batched: 32 x [62x62]@[62,in_f])
    launch_gemm(stream, adjm, xin, adjx, nullptr, nullptr, nullptr, nullptr,
                62, in_f, 62, ACT_NONE, 32, 3844, (long long)62 * in_f, (long long)62 * in_f);
    // conv0: h = relu(adjx rel_w + rel_b + x root_w)
    launch_gemm(stream, adjx, SG(g, 3), tbuf, SG(g, 2), nullptr, nullptr, nullptr,
                1984, 64, in_f, ACT_NONE);
    launch_gemm(stream, xin, SG(g, 4), hbuf, nullptr, tbuf, nullptr, nullptr,
                1984, 64, in_f, ACT_RELU);
    float* h = hbuf; float* h2 = h2buf;
    for (int j = 1; j <= 5; ++j) {
      launch_gemm(stream, adjm, h, adjx, nullptr, nullptr, nullptr, nullptr,
                  62, 64, 62, ACT_NONE, 32, 3844, 62 * 64, 62 * 64); // adjh = adj @ h
      launch_gemm(stream, adjx, SG(g, 3 + 3 * j), tbuf, SG(g, 2 + 3 * j),
                  nullptr, nullptr, nullptr, 1984, 64, 64, ACT_NONE);
      launch_gemm(stream, h, SG(g, 4 + 3 * j), h2, nullptr, tbuf, nullptr, h,
                  1984, 64, 64, ACT_RELU);                           // h' = h + relu(...)
      float* tmp = h; h = h2; h2 = tmp;
    }
    // conv6 (64->32), then * x_mask
    launch_gemm(stream, adjm, h, adjx, nullptr, nullptr, nullptr, nullptr,
                62, 64, 62, ACT_NONE, 32, 3844, 62 * 64, 62 * 64);
    launch_gemm(stream, adjx, SG(g, 21), tbuf, SG(g, 20), nullptr, nullptr, nullptr,
                1984, 32, 64, ACT_NONE);
    launch_gemm(stream, h, SG(g, 22), out_slice, nullptr, tbuf, gmask, nullptr,
                1984, 32, 64, ACT_RELU);
  };

  // ================= phase 1: MHSA band attention =================
  launch_gemm(stream, x, wq, q,  nullptr, nullptr, nullptr, nullptr, 9920, 512, 265, ACT_NONE);
  launch_gemm(stream, x, wk, kb, nullptr, nullptr, nullptr, nullptr, 9920, 512, 265, ACT_NONE);
  launch_gemm(stream, x, wv, vb, nullptr, nullptr, nullptr, nullptr, 9920, 512, 265, ACT_NONE);
  { int t = 1984 * 8 * 5;
    k_attn_scores<<<(t + 127) / 128, 128, 0, stream>>>(q, kb, dist, t); }
  { int t = 1984 * 8 * 5 * 64;
    k_attn_apply<<<(t + 255) / 256, 256, 0, stream>>>(dist, vb, att, t); }
  launch_gemm(stream, att, wo, sb, bo, nullptr, nullptr, nullptr, 1984, 5, 2560, ACT_NONE);
  k_band<<<(1984 + 127) / 128, 128, 0, stream>>>(sb, cwa_w, cwa_b, wband, 1984);
  { int t = 2628800;
    k_scale<<<(t + 255) / 256, 256, 0, stream>>>(x, wband, xs, t); }

  // ================= phase 2: conv frontend + graph blocks =================
  { long long t = (long long)1984 * 32 * 5 * 66;   // mc1 3x3 pad1 + pool4
    k_convpool<<<(unsigned)((t + 255) / 256), 256, 0, stream>>>(
        xs, mc1_w, mc1_b, p1, 1984, 1, 32, 5, 265, 3, 3, 1, 1, 265, 4, 0, 66); }
  graph(p1, 10560, 0, xk + 0 * 63488);
  { long long t = (long long)1984 * 32 * 5 * 17;   // mc2 3x3 pad1 + pool4 pad2
    k_convpool<<<(unsigned)((t + 255) / 256), 256, 0, stream>>>(
        p1, mc2_w, mc2_b, p2, 1984, 32, 32, 5, 66, 3, 3, 1, 1, 66, 4, 2, 17); }
  graph(p2, 2720, 1, xk + 1 * 63488);
  { long long t = (long long)1984 * 32 * 5 * 7;    // c1 1x3 + pool2
    k_convpool<<<(unsigned)((t + 255) / 256), 256, 0, stream>>>(
        p2, c1_w, c1_b, p3, 1984, 32, 32, 5, 17, 1, 3, 0, 0, 15, 2, 0, 7); }
  graph(p3, 1120, 2, xk + 2 * 63488);
  { long long t = (long long)1984 * 64 * 5 * 2;    // c2 1x3 + pool2
    k_convpool<<<(unsigned)((t + 255) / 256), 256, 0, stream>>>(
        p3, c2_w, c2_b, p4, 1984, 32, 64, 5, 7, 1, 3, 0, 0, 5, 2, 0, 2); }
  graph(p4, 640, 3, xk + 3 * 63488);
  { long long t = (long long)1984 * 128 * 5;       // c3 1x2, no pool
    k_convpool<<<(unsigned)((t + 255) / 256), 256, 0, stream>>>(
        p4, c3_w, c3_b, p5, 1984, 64, 128, 5, 2, 1, 2, 0, 0, 1, 1, 0, 1); }
  graph(p5, 640, 4, xk + 4 * 63488);

  // ================= phase 3: gated head =================
  { int t = 317440;
    k_concat<<<(t + 255) / 256, 256, 0, stream>>>(xk, cat, t); }
  launch_gemm(stream, cat, gate_w, gated, gate_b, nullptr, cat, nullptr,
              32, 9920, 9920, ACT_TANH);                  // tanh(cat W + b) * cat
  launch_gemm(stream, gated, lin1_w, l1, lin1_b, nullptr, nullptr, nullptr,
              32, 1024, 9920, ACT_RELU);
  launch_gemm(stream, l1, lin2_w, (float*)d_out, lin2_b, nullptr, nullptr, nullptr,
              32, 3, 1024, ACT_NONE);
}
